// SA_FF_Layer_22093311770698
// MI455X (gfx1250) — matrix operand order repaired and checked by reading, not compiled, not run on hardware
//
#include <hip/hip_runtime.h>
#include <math.h>

// ---------------------------------------------------------------------------
// CDNA5 (gfx1250) transformer block: LN -> QKV GEMM -> flash attention ->
// out-proj+residual -> LN -> FF1+GELU -> FF2+residual.
// All GEMMs via v_wmma_f32_16x16x32_bf16 (fp32 accumulate), wave32.
// ---------------------------------------------------------------------------

typedef __bf16 bf16_t;
typedef __attribute__((ext_vector_type(16))) __bf16 v16bf;
typedef __attribute__((ext_vector_type(8)))  __bf16 bf16x8;
typedef __attribute__((ext_vector_type(8)))  float   v8f;
typedef __attribute__((ext_vector_type(4)))  float   f32x4;

#define LN_EPS 1e-5f

__device__ __forceinline__ v16bf cat16(bf16x8 lo, bf16x8 hi) {
  return __builtin_shufflevector(lo, hi, 0, 1, 2, 3, 4, 5, 6, 7,
                                 8, 9, 10, 11, 12, 13, 14, 15);
}

__device__ __forceinline__ v8f wmma_bf16(v16bf a, v16bf b, v8f c) {
  // (neg_a, A, neg_b, B, c_mod, C, reuse_a, reuse_b)
  return __builtin_amdgcn_wmma_f32_16x16x32_bf16(false, a, false, b,
                                                 (short)0, c, false, false);
}

// ---------------------------------------------------------------------------
// fp32 -> bf16 conversion (weights), vectorized x4
// ---------------------------------------------------------------------------
__global__ __launch_bounds__(256)
void f32_to_bf16_kernel(const float* __restrict__ in, bf16_t* __restrict__ out,
                        size_t n4) {
  size_t i = (size_t)blockIdx.x * blockDim.x + threadIdx.x;
  if (i < n4) {
    f32x4 v = *(const f32x4*)(in + i * 4);
    out[i * 4 + 0] = (bf16_t)v.x;
    out[i * 4 + 1] = (bf16_t)v.y;
    out[i * 4 + 2] = (bf16_t)v.z;
    out[i * 4 + 3] = (bf16_t)v.w;
  }
}

// ---------------------------------------------------------------------------
// LayerNorm over D=1024: one row per 256-thread block (8 wave32).
// Writes fp32 (residual path) and bf16 (WMMA operand path).
// ---------------------------------------------------------------------------
__global__ __launch_bounds__(256)
void layernorm_kernel(const float* __restrict__ in, const float* __restrict__ g,
                      const float* __restrict__ b, float* __restrict__ out_f32,
                      bf16_t* __restrict__ out_b16, int D) {
  __shared__ float red[8];
  const int t = threadIdx.x, lane = t & 31, wave = t >> 5;
  const size_t row = blockIdx.x;
  const float* x = in + row * (size_t)D;
  const int c0 = t * 4;

  f32x4 v = *(const f32x4*)(x + c0);
  float s = v.x + v.y + v.z + v.w;
#pragma unroll
  for (int o = 16; o > 0; o >>= 1) s += __shfl_xor(s, o, 32);
  if (lane == 0) red[wave] = s;
  __syncthreads();
  float tot = 0.f;
#pragma unroll
  for (int i = 0; i < 8; ++i) tot += red[i];
  const float mu = tot / (float)D;
  __syncthreads();

  f32x4 d;
  d.x = v.x - mu; d.y = v.y - mu; d.z = v.z - mu; d.w = v.w - mu;
  float s2 = d.x * d.x + d.y * d.y + d.z * d.z + d.w * d.w;
#pragma unroll
  for (int o = 16; o > 0; o >>= 1) s2 += __shfl_xor(s2, o, 32);
  if (lane == 0) red[wave] = s2;
  __syncthreads();
  float tot2 = 0.f;
#pragma unroll
  for (int i = 0; i < 8; ++i) tot2 += red[i];
  const float rs = rsqrtf(tot2 / (float)D + LN_EPS);

  f32x4 gg = *(const f32x4*)(g + c0);
  f32x4 bb = *(const f32x4*)(b + c0);
  f32x4 y;
  y.x = d.x * rs * gg.x + bb.x;
  y.y = d.y * rs * gg.y + bb.y;
  y.z = d.z * rs * gg.z + bb.z;
  y.w = d.w * rs * gg.w + bb.w;
  *(f32x4*)(out_f32 + row * (size_t)D + c0) = y;
  bf16_t* ob = out_b16 + row * (size_t)D + c0;
  ob[0] = (bf16_t)y.x; ob[1] = (bf16_t)y.y;
  ob[2] = (bf16_t)y.z; ob[3] = (bf16_t)y.w;
}

// ---------------------------------------------------------------------------
// Tiled bf16 WMMA GEMM: C(MxN) = A(MxK) * B(KxN), row-major, fp32 accum.
// BM=BN=128, BK=32; 256 threads = 8 wave32 arranged 2(M) x 4(N);
// each wave computes 64x32 = 4x2 WMMA tiles. Double-buffered LDS.
// Epilogues: 0 = store bf16; 1 = +residual, store f32; 2 = +bias, GELU, bf16.
// ---------------------------------------------------------------------------
enum { EPI_STORE_BF16 = 0, EPI_RESID_F32 = 1, EPI_BIAS_GELU_BF16 = 2 };

template <int EPI>
__global__ __launch_bounds__(256)
void gemm_bf16_kernel(const bf16_t* __restrict__ A, const bf16_t* __restrict__ B,
                      void* __restrict__ Cout, const float* __restrict__ bias,
                      const float* __restrict__ resid, int M, int N, int K) {
  constexpr int BM = 128, BN = 128, BK = 32, PAD = 8;
  __shared__ bf16_t Asm[2][BM][BK + PAD];  // 20.0 KB
  __shared__ bf16_t Bsm[2][BN][BK + PAD];  // transposed [n][k], 20.0 KB

  const int t = threadIdx.x;
  const int lane = t & 31, wave = t >> 5;
  const int half = lane >> 4, l16 = lane & 15;
  const int waveM = wave & 1;   // 0..1 -> 64-row slab
  const int waveN = wave >> 1;  // 0..3 -> 32-col slab
  const size_t bm = (size_t)blockIdx.y * BM;
  const size_t bn = (size_t)blockIdx.x * BN;

  const int arow = t >> 1, acol = (t & 1) * 16;   // A: 128 rows x 32 cols
  const int bkr = t >> 3, bnc = (t & 7) * 16;     // B: 32 rows x 128 cols

  v8f acc[4][2] = {};

  auto load_tile = [&](int kt, int buf) {
    const bf16_t* ag = A + (bm + arow) * (size_t)K + (size_t)kt * BK + acol;
    bf16x8 a0 = *(const bf16x8*)(ag);
    bf16x8 a1 = *(const bf16x8*)(ag + 8);
    *(bf16x8*)(&Asm[buf][arow][acol]) = a0;
    *(bf16x8*)(&Asm[buf][arow][acol + 8]) = a1;
    const bf16_t* bg = B + ((size_t)kt * BK + bkr) * (size_t)N + bn + bnc;
    bf16x8 b0 = *(const bf16x8*)(bg);
    bf16x8 b1 = *(const bf16x8*)(bg + 8);
#pragma unroll
    for (int i = 0; i < 8; ++i) Bsm[buf][bnc + i][bkr] = b0[i];
#pragma unroll
    for (int i = 0; i < 8; ++i) Bsm[buf][bnc + 8 + i][bkr] = b1[i];
  };

  const int KT = K / BK;
  load_tile(0, 0);
  __syncthreads();

  int buf = 0;
  for (int kt = 0; kt < KT; ++kt) {
    if (kt + 2 < KT) {  // L2 prefetch of the tile after next
      __builtin_prefetch(A + (bm + arow) * (size_t)K + (size_t)(kt + 2) * BK + acol, 0, 0);
      __builtin_prefetch(B + ((size_t)(kt + 2) * BK + bkr) * (size_t)N + bn + bnc, 0, 0);
    }
    if (kt + 1 < KT) load_tile(kt + 1, buf ^ 1);

    v16bf af[4], bfrag[2];
#pragma unroll
    for (int mt = 0; mt < 4; ++mt) {
      // A frag: m = lane%16, K = (j<8 ? j : j+8) + 8*(lane/16)
      const bf16_t* r = &Asm[buf][waveM * 64 + mt * 16 + l16][0];
      af[mt] = cat16(*(const bf16x8*)(r + half * 8),
                     *(const bf16x8*)(r + 16 + half * 8));
    }
#pragma unroll
    for (int nt = 0; nt < 2; ++nt) {
      // B frag: N = lane%16, K = (j<8 ? j : j+8) + 8*(lane/16)
      const bf16_t* r = &Bsm[buf][waveN * 32 + nt * 16 + l16][0];
      bfrag[nt] = cat16(*(const bf16x8*)(r + half * 8),
                        *(const bf16x8*)(r + 16 + half * 8));
    }
#pragma unroll
    for (int mt = 0; mt < 4; ++mt)
#pragma unroll
      for (int nt = 0; nt < 2; ++nt)
        acc[mt][nt] = wmma_bf16(af[mt], bfrag[nt], acc[mt][nt]);

    __syncthreads();
    buf ^= 1;
  }

#pragma unroll
  for (int mt = 0; mt < 4; ++mt) {
#pragma unroll
    for (int nt = 0; nt < 2; ++nt) {
#pragma unroll
      for (int r = 0; r < 8; ++r) {
        // C layout: row = r + 8*(lane/16), col = lane%16
        size_t row = bm + waveM * 64 + mt * 16 + half * 8 + r;
        size_t col = bn + waveN * 32 + nt * 16 + l16;
        size_t idx = row * (size_t)N + col;
        float v = acc[mt][nt][r];
        if constexpr (EPI == EPI_STORE_BF16) {
          ((bf16_t*)Cout)[idx] = (bf16_t)v;
        } else if constexpr (EPI == EPI_RESID_F32) {
          ((float*)Cout)[idx] = v + resid[idx];
        } else {
          v += bias[col];
          v = 0.5f * v * (1.0f + erff(v * 0.70710678118654752f));  // exact GELU
          ((bf16_t*)Cout)[idx] = (bf16_t)v;
        }
      }
    }
  }
}

// ---------------------------------------------------------------------------
// Flash attention, head dim 64, bf16 WMMA, fp32 online softmax.
// Block: 128 threads (4 wave32), 64 queries; each wave owns 16 queries.
//
// TRANSPOSED formulation: S^T = K * Q^T puts KEYS on the WMMA row axis
// (in-lane across accumulator VGPRs + the lane/lane^16 halves) and QUERIES
// on the column axis (lane%16). The softmax reduction over keys is then an
// in-lane VALU tree + ONE __shfl_xor(16) -- no ds_bpermute butterflies.
// O^T = V^T * P^T keeps everything in WMMA. P^T relayout packs as b128.
// KV processed in 64-key chunks staged in LDS (V transposed).
// qkv layout: [B,S,3,H,64] row-major (direct output of the QKV GEMM).
// ---------------------------------------------------------------------------
__global__ __launch_bounds__(128)
void flash_attn_kernel(const bf16_t* __restrict__ qkv, bf16_t* __restrict__ O,
                       int S, int H) {
  constexpr int AD = 64;
  __shared__ bf16_t Ksm[64][AD + 8];       // [key][dim]       9.0 KB
  __shared__ bf16_t Vsm[AD][64 + 8];       // [dim][key]       9.0 KB
  __shared__ bf16_t Psm[4][16][64 + 8];    // [wave][query][key] 9.0 KB

  const int t = threadIdx.x, lane = t & 31, wave = t >> 5;
  const int half = lane >> 4, l16 = lane & 15;
  const int b = blockIdx.z, h = blockIdx.y;
  const int q0 = blockIdx.x * 64;
  const int D3 = 3 * H * AD;  // 3072
  const int Dd = H * AD;      // 1024
  const size_t bS = (size_t)b * S;

  // Q as B-fragments of Q^T (dims x queries): query = lane%16,
  // dim = kk*32 + j + 16*(lane/16) -> contiguous global b128 pairs.
  v16bf qb[2];
  const bf16_t* qr = qkv + (bS + q0 + wave * 16 + l16) * (size_t)D3 + h * AD;
#pragma unroll
  for (int kk = 0; kk < 2; ++kk)
    qb[kk] = cat16(*(const bf16x8*)(qr + kk * 32 + half * 16),
                   *(const bf16x8*)(qr + kk * 32 + half * 16 + 8));

  float mcur = -1e30f, lcur = 0.f;   // per-query (per-lane) softmax stats
  v8f oacc[4] = {};                  // O^T: 64 dims x 16 queries

  const int key = t >> 1;          // 0..63
  const int dseg = (t & 1) * 32;   // 0 or 32

  for (int kv0 = 0; kv0 < S; kv0 += 64) {
    // Stage K (row-major [key][dim]) and V (transposed [dim][key]) into LDS.
    const bf16_t* kg = qkv + (bS + kv0 + key) * (size_t)D3 + Dd + h * AD + dseg;
    bf16x8 kr[4];
#pragma unroll
    for (int i = 0; i < 4; ++i) kr[i] = *(const bf16x8*)(kg + i * 8);
#pragma unroll
    for (int i = 0; i < 4; ++i) *(bf16x8*)(&Ksm[key][dseg + i * 8]) = kr[i];
    const bf16_t* vg =
        qkv + (bS + kv0 + key) * (size_t)D3 + 2 * Dd + h * AD + dseg;
    bf16x8 vr[4];
#pragma unroll
    for (int i = 0; i < 4; ++i) vr[i] = *(const bf16x8*)(vg + i * 8);
#pragma unroll
    for (int i = 0; i < 4; ++i)
#pragma unroll
      for (int j = 0; j < 8; ++j) Vsm[dseg + i * 8 + j][key] = vr[i][j];
    __syncthreads();

    // S^T = K * Q^T : 4 key-tiles of 16; A = K rows, B = Q^T.
    v8f sc[4];
#pragma unroll
    for (int kt2 = 0; kt2 < 4; ++kt2) {
      v8f c = {};
#pragma unroll
      for (int kk = 0; kk < 2; ++kk) {
        const bf16_t* ar = &Ksm[kt2 * 16 + l16][kk * 32];
        c = wmma_bf16(cat16(*(const bf16x8*)(ar + half * 16),
                            *(const bf16x8*)(ar + half * 16 + 8)),
                      qb[kk], c);
      }
      sc[kt2] = c * 0.125f;  // 1/sqrt(64)
    }

    // Online softmax per query column: in-lane tree + one xor-16 shuffle.
    float mx = sc[0][0];
#pragma unroll
    for (int kt2 = 0; kt2 < 4; ++kt2)
#pragma unroll
      for (int r = 0; r < 8; ++r) mx = fmaxf(mx, sc[kt2][r]);
    mx = fmaxf(mx, __shfl_xor(mx, 16, 32));
    const float mnew = fmaxf(mcur, mx);
    const float scale = __expf(mcur - mnew);
    float rsum = 0.f;
#pragma unroll
    for (int kt2 = 0; kt2 < 4; ++kt2) {
      bf16x8 pk;
#pragma unroll
      for (int r = 0; r < 8; ++r) {
        const float p = __expf(sc[kt2][r] - mnew);
        rsum += p;
        pk[r] = (bf16_t)p;
      }
      // P^T to LDS [query][key]; 8 keys per b128 store, halves disjoint.
      *(bf16x8*)(&Psm[wave][l16][kt2 * 16 + half * 8]) = pk;
    }
    rsum += __shfl_xor(rsum, 16, 32);
    lcur = lcur * scale + rsum;
    mcur = mnew;
#pragma unroll
    for (int dd = 0; dd < 4; ++dd) oacc[dd] *= scale;
    __syncthreads();

    // O^T += V^T * P^T : A = V^T (dims x keys), B = P^T (keys x queries).
#pragma unroll
    for (int ks = 0; ks < 2; ++ks) {
      const bf16_t* pr = &Psm[wave][l16][ks * 32 + half * 16];
      v16bf pb = cat16(*(const bf16x8*)pr, *(const bf16x8*)(pr + 8));
#pragma unroll
      for (int dd = 0; dd < 4; ++dd) {
        const bf16_t* vv = &Vsm[dd * 16 + l16][ks * 32];
        oacc[dd] = wmma_bf16(cat16(*(const bf16x8*)(vv + half * 16),
                                   *(const bf16x8*)(vv + half * 16 + 8)),
                             pb, oacc[dd]);
      }
    }
    __syncthreads();
  }

  // Normalize and store O^T back as [B*S, D] bf16 (feeds out-proj GEMM).
  const float inv = 1.f / lcur;
  const size_t qrow = bS + q0 + wave * 16 + l16;
#pragma unroll
  for (int dd = 0; dd < 4; ++dd)
#pragma unroll
    for (int r = 0; r < 8; ++r)
      O[qrow * (size_t)Dd + h * AD + dd * 16 + half * 8 + r] =
          (bf16_t)(oacc[dd][r] * inv);
}

// ---------------------------------------------------------------------------
// Host-side orchestration
// ---------------------------------------------------------------------------
extern "C" void kernel_launch(void* const* d_in, const int* in_sizes, int n_in,
                              void* d_out, int out_size, void* d_ws,
                              size_t ws_size, hipStream_t stream) {
  const int B = 4, S = 2048, D = 1024, F = 4096, H = 16;
  const int R = B * S;  // 8192 rows

  const float* inputs = (const float*)d_in[0];
  const float* ln1_g = (const float*)d_in[1];
  const float* ln1_b = (const float*)d_in[2];
  const float* w_qkv = (const float*)d_in[3];
  const float* w_out = (const float*)d_in[4];
  const float* ff_g = (const float*)d_in[5];
  const float* ff_b = (const float*)d_in[6];
  const float* ff_w1 = (const float*)d_in[7];
  const float* ff_b1 = (const float*)d_in[8];
  const float* ff_w2 = (const float*)d_in[9];

  char* ws = (char*)d_ws;
  size_t off = 0;
  auto alloc = [&](size_t bytes) -> void* {
    void* p = ws + off;
    off += (bytes + 255) & ~(size_t)255;
    return p;
  };
  bf16_t* xb = (bf16_t*)alloc((size_t)R * D * 2);      // LN1 bf16 (later: yb)
  float* x32 = (float*)alloc((size_t)R * D * 4);       // LN1 f32  (later: y32)
  bf16_t* wqkv_b = (bf16_t*)alloc((size_t)D * 3 * D * 2);
  bf16_t* wout_b = (bf16_t*)alloc((size_t)D * D * 2);
  bf16_t* w1_b = (bf16_t*)alloc((size_t)D * F * 2);
  bf16_t* w2_b = (bf16_t*)alloc((size_t)F * D * 2);
  bf16_t* qkv_b = (bf16_t*)alloc((size_t)R * 3 * D * 2);  // 48 MiB
  bf16_t* o_b = (bf16_t*)alloc((size_t)R * D * 2);        // 16 MiB
  float* ctx32 = (float*)alloc((size_t)R * D * 4);
  // Aliases (lifetimes are disjoint along the stream order):
  bf16_t* yb = xb;
  float* y32 = x32;
  bf16_t* hb = qkv_b;  // 8192*4096*2 = 64 MiB == qkv_b(48) + o_b(16) exactly

  auto cvt = [&](const float* src, bf16_t* dst, size_t n) {
    size_t n4 = n / 4;
    f32_to_bf16_kernel<<<dim3((unsigned)((n4 + 255) / 256)), dim3(256), 0,
                         stream>>>(src, dst, n4);
  };
  cvt(w_qkv, wqkv_b, (size_t)D * 3 * D);
  cvt(w_out, wout_b, (size_t)D * D);
  cvt(ff_w1, w1_b, (size_t)D * F);
  cvt(ff_w2, w2_b, (size_t)F * D);

  // x = LN(inputs)
  layernorm_kernel<<<dim3(R), dim3(256), 0, stream>>>(inputs, ln1_g, ln1_b,
                                                      x32, xb, D);
  // qkv = x @ w_qkv  (8192 x 3072 x 1024)
  gemm_bf16_kernel<EPI_STORE_BF16>
      <<<dim3(3 * D / 128, R / 128), dim3(256), 0, stream>>>(
          xb, wqkv_b, qkv_b, nullptr, nullptr, R, 3 * D, D);
  // o = softmax(q k^T / 8) v   per (b, h)
  flash_attn_kernel<<<dim3(S / 64, H, B), dim3(128), 0, stream>>>(qkv_b, o_b,
                                                                  S, H);
  // context = o @ w_out + x
  gemm_bf16_kernel<EPI_RESID_F32>
      <<<dim3(D / 128, R / 128), dim3(256), 0, stream>>>(
          o_b, wout_b, ctx32, nullptr, x32, R, D, D);
  // y = LN(context)
  layernorm_kernel<<<dim3(R), dim3(256), 0, stream>>>(ctx32, ff_g, ff_b, y32,
                                                      yb, D);
  // h = gelu(y @ ff_w1 + b1)   (8192 x 4096 x 1024)
  gemm_bf16_kernel<EPI_BIAS_GELU_BF16>
      <<<dim3(F / 128, R / 128), dim3(256), 0, stream>>>(
          yb, w1_b, hb, ff_b1, nullptr, R, F, D);
  // out = h @ ff_w2 + y        (8192 x 1024 x 4096)
  gemm_bf16_kernel<EPI_RESID_F32>
      <<<dim3(D / 128, R / 128), dim3(256), 0, stream>>>(
          hb, w2_b, (float*)d_out, nullptr, y32, R, D, F);

  (void)in_sizes; (void)n_in; (void)out_size; (void)ws_size;
}
